// MultiHeadAttention_79723182949055
// MI455X (gfx1250) — matrix operand order repaired and checked by reading, not compiled, not run on hardware
//
#include <hip/hip_runtime.h>

// ---------------------------------------------------------------------------
// MultiHeadAttention (bilinear, softmax over keys) for MI455X / gfx1250.
// Pipeline: 3x projection GEMM (f32 -> bf16 ws) -> flash attention -> out GEMM.
// Matrix math: v_wmma_f32_16x16x32_bf16 (wave32). LDS tiles double-buffered;
// attention K-tiles streamed with GLOBAL_LOAD_ASYNC_TO_LDS_B128 (ASYNCcnt).
// ---------------------------------------------------------------------------

typedef __attribute__((ext_vector_type(16))) __bf16 v16bf;
typedef __attribute__((ext_vector_type(8)))  float  v8f;
typedef __attribute__((ext_vector_type(4)))  int    v4i;

#define NROWS 4096   // B*N
#define CDIM  1024   // K*D == P == Q
#define SEQ   2048
#define NB    2
#define NHEAD 16
#define HD    64
#define INVT  0.125f // 1/sqrt(D)=1/8

__device__ __forceinline__ v8f wmma_bf16(v16bf a, v16bf b, v8f c) {
  // D = A(16x32) * B(32x16) + C, f32 accumulate
  return __builtin_amdgcn_wmma_f32_16x16x32_bf16(false, a, false, b,
                                                 (short)0, c, false, false);
}

// 16-byte global->LDS copy: async DMA path on gfx1250 (ASYNCcnt-tracked),
// synchronous fallback otherwise. Builtin takes v4i pointers (per diagnostic).
__device__ __forceinline__ void copy16_g2l(const __bf16* gsrc, __bf16* ldst) {
#if __has_builtin(__builtin_amdgcn_global_load_async_to_lds_b128)
  __builtin_amdgcn_global_load_async_to_lds_b128(
      (__attribute__((address_space(1))) v4i*)(v4i*)gsrc,
      (__attribute__((address_space(3))) v4i*)(v4i*)ldst, 0, 0);
#else
  *(uint4*)ldst = *(const uint4*)gsrc;
#endif
}

__device__ __forceinline__ void wait_async0() {
#if __has_builtin(__builtin_amdgcn_global_load_async_to_lds_b128)
#if __has_builtin(__builtin_amdgcn_s_wait_asynccnt)
  __builtin_amdgcn_s_wait_asynccnt(0);
#else
  asm volatile("s_wait_asynccnt 0x0" ::: "memory");
#endif
#endif
}

// ---------------------------------------------------------------------------
// GEMM: C[4096 x 1024] = A[4096 x 1024] * W
// WSWAP==0 (projection): W element (contraction p, out col c) = Wt[c>>6][p][c&63]
// WSWAP==1 (output):     W element (contraction c, out col q) = Wt[c>>6][q][c&63]
// Block: 256 threads / 8 waves. Block tile 128 rows x 64 cols; wave = 16 rows.
// Weight chunk (32 x 64) -> LDS [col][k] bf16, stride-36 pad, double-buffered:
// next chunk's global f32 loads are issued before this chunk's WMMAs.
// ---------------------------------------------------------------------------
template<int WSWAP, typename AT, typename OT>
__global__ __launch_bounds__(256)
void gemm1024_wmma(const AT* __restrict__ A, const float* __restrict__ W,
                   OT* __restrict__ C)
{
  __shared__ __bf16 wlds[2][64 * 36];
  const int tid  = threadIdx.x;
  const int wave = tid >> 5, lane = tid & 31;
  const int half = lane >> 4, ln = lane & 15;
  const int row0 = blockIdx.x * 128;
  const int col0 = blockIdx.y * 64;

  // per-thread staging map: 8 (i = contraction-local, cl = col-local) pairs
  auto wload = [&](int kk, float* wreg) {
    if (WSWAP == 0) {
      const int kh = col0 >> 6;                         // col tile in one head
      const float* wp = W + ((size_t)kh * 1024 + kk) * 64;
#pragma unroll
      for (int r = 0; r < 8; ++r) {
        const int s = tid + 256 * r, i = s >> 6, cl = s & 63;  // coalesced in cl
        wreg[r] = wp[(size_t)i * 64 + cl];
      }
    } else {
      const int kh = kk >> 6, d0 = kk & 63;             // chunk in one head
      const float* wp = W + ((size_t)kh * 1024 + col0) * 64 + d0;
#pragma unroll
      for (int r = 0; r < 8; ++r) {
        const int s = tid + 256 * r, i = s & 31, cl = s >> 5;  // coalesced in i
        wreg[r] = wp[(size_t)cl * 64 + i];
      }
    }
  };
  auto wstore = [&](int p, const float* wreg) {
#pragma unroll
    for (int r = 0; r < 8; ++r) {
      const int s = tid + 256 * r;
      const int i  = (WSWAP == 0) ? (s >> 6) : (s & 31);
      const int cl = (WSWAP == 0) ? (s & 63) : (s >> 5);
      wlds[p][cl * 36 + i] = (__bf16)wreg[r];
    }
  };

  float wreg[8];
  wload(0, wreg);
  wstore(0, wreg);
  __syncthreads();

  v8f acc[4];
#pragma unroll
  for (int g = 0; g < 4; ++g) acc[g] = (v8f){0.f,0.f,0.f,0.f,0.f,0.f,0.f,0.f};

  for (int kk = 0; kk < CDIM; kk += 32) {
    const int  p   = (kk >> 5) & 1;
    const bool nxt = (kk + 32) < CDIM;
    if (nxt) wload(kk + 32, wreg);       // prefetch next chunk (overlaps WMMA)

    // A fragment: row = row0 + wave*16 + ln; one 16-elem contiguous run at 16*half
    const AT* arow = A + (size_t)(row0 + wave * 16 + ln) * CDIM + kk;
    v16bf af;
#pragma unroll
    for (int e = 0; e < 8; ++e) af[e]     = (__bf16)arow[16 * half + e];
#pragma unroll
    for (int e = 0; e < 8; ++e) af[8 + e] = (__bf16)arow[16 * half + 8 + e];

#pragma unroll
    for (int g = 0; g < 4; ++g) {
      // B fragment: col = col0 + g*16 + ln; K-row = 16*half + e
      const __bf16* wl = &wlds[p][(g * 16 + ln) * 36 + 16 * half];
      v16bf bf;
#pragma unroll
      for (int e = 0; e < 16; ++e) bf[e] = wl[e];
      acc[g] = wmma_bf16(af, bf, acc[g]);
    }

    if (nxt) wstore(p ^ 1, wreg);
    __syncthreads();
  }

  // store: D element j -> row 8*half+j, col ln (within each 16x16 tile)
#pragma unroll
  for (int g = 0; g < 4; ++g) {
#pragma unroll
    for (int j = 0; j < 8; ++j) {
      const int r = row0 + wave * 16 + 8 * half + j;
      const int c = col0 + g * 16 + ln;
      C[(size_t)r * CDIM + c] = (OT)acc[g][j];
    }
  }
}

// ---------------------------------------------------------------------------
// Flash attention per (head k, batch b, 64-query tile). Scores computed
// TRANSPOSED: T[m,n] = sum_d K[m,d]*Q[n,d], so softmax over m is (nearly)
// lane-local and exp(T) lands directly in the A-frag layout for P@V.
// Block: 128 threads / 4 waves; each wave owns 16 queries, loops 2048 keys
// in 32-key steps. K tile: async global->LDS (pure bf16 copy), V tile:
// register-staged transpose. Both double-buffered; one barrier per step.
// ---------------------------------------------------------------------------
__global__ __launch_bounds__(128)
void flash_attn_wmma(const __bf16* __restrict__ Rk,  // keys    [4096][1024]
                     const __bf16* __restrict__ Sq,  // queries [4096][1024]
                     const __bf16* __restrict__ Vv,  // values  [4096][1024]
                     const float*  __restrict__ mask, // [2048][2048], [m][n]
                     __bf16* __restrict__ O)          // [4096][1024]
{
  __shared__ __align__(16) __bf16 klds[2][32 * 72];  // [m_local][d], pad 72
  __shared__ __align__(16) __bf16 vlds[2][64 * 36];  // [d][m_local], pad 36

  const int tid  = threadIdx.x;
  const int wave = tid >> 5, lane = tid & 31;
  const int half = lane >> 4, ln = lane & 15;
  const int k = blockIdx.y, b = blockIdx.z;
  const int q0 = blockIdx.x * 64;
  const int nq = q0 + wave * 16 + ln;     // query index tracked by this lane

  const __bf16* kbase = Rk + (size_t)b * SEQ * CDIM + k * HD;
  const __bf16* vbase = Vv + (size_t)b * SEQ * CDIM + k * HD;

  // K tile: 32 rows x 64 bf16 = 256 x 16B chunks, 2 per thread, async DMA
  auto kstage = [&](int p, int mb) {
#pragma unroll
    for (int r = 0; r < 2; ++r) {
      const int ci = tid + 128 * r;           // chunk id
      const int mm = ci >> 2, part = ci & 3;  // 4 x 16B per key row
      copy16_g2l(kbase + (size_t)(mb + mm) * CDIM + part * 8,
                 &klds[p][mm * 72 + part * 8]);
    }
  };
  // V tile: coalesced global loads to regs (issued early), transposed LDS store
  auto vload = [&](int mb, __bf16* vreg) {
#pragma unroll
    for (int r = 0; r < 16; ++r) {
      const int s = tid + 128 * r, mm = s >> 6, d = s & 63;
      vreg[r] = vbase[(size_t)(mb + mm) * CDIM + d];
    }
  };
  auto vstore = [&](int p, const __bf16* vreg) {
#pragma unroll
    for (int r = 0; r < 16; ++r) {
      const int s = tid + 128 * r, mm = s >> 6, d = s & 63;
      vlds[p][d * 36 + mm] = vreg[r];
    }
  };

  // Q fragments (loop invariant): B-frag element e -> d = c*32 + 16*half + e
  const __bf16* qrow = Sq + (size_t)(b * SEQ + nq) * CDIM + k * HD;
  v16bf qf[2];
#pragma unroll
  for (int c = 0; c < 2; ++c)
#pragma unroll
    for (int e = 0; e < 16; ++e) qf[c][e] = qrow[c * 32 + 16 * half + e];

  v8f acc[4];
#pragma unroll
  for (int g = 0; g < 4; ++g) acc[g] = (v8f){0.f,0.f,0.f,0.f,0.f,0.f,0.f,0.f};
  float mrun = -1e30f, lrun = 0.f;

  // prologue: fill buffer 0
  __bf16 vreg[16];
  kstage(0, 0);
  vload(0, vreg);
  vstore(0, vreg);
  wait_async0();
  __syncthreads();

  for (int mb = 0; mb < SEQ; mb += 32) {
    const int  p   = (mb >> 5) & 1;
    const bool nxt = (mb + 32) < SEQ;
    if (nxt) {                      // stream next tile while computing this one
      kstage(p ^ 1, mb + 32);       // async DMA into other buffer
      vload(mb + 32, vreg);         // global loads in flight
    }

    // ---- scores transposed: two 16x16 tiles (m-sub 0/1), k-split over d ----
    float lg[2][8];
#pragma unroll
    for (int ms = 0; ms < 2; ++ms) {
      v8f sc = (v8f){0.f,0.f,0.f,0.f,0.f,0.f,0.f,0.f};
#pragma unroll
      for (int c = 0; c < 2; ++c) {
        const __bf16* kl = &klds[p][(ms * 16 + ln) * 72 + c * 32];
        v16bf kf;
#pragma unroll
        for (int e = 0; e < 8; ++e) kf[e]     = kl[16 * half + e];
#pragma unroll
        for (int e = 0; e < 8; ++e) kf[8 + e] = kl[16 * half + 8 + e];
        sc = wmma_bf16(kf, qf[c], sc);
      }
      // lane holds col n = nq, rows m = mb + ms*16 + 8*half + j
#pragma unroll
      for (int j = 0; j < 8; ++j) {
        const int mg = mb + ms * 16 + 8 * half + j;
        lg[ms][j] = (sc[j] + mask[(size_t)mg * SEQ + nq]) * INVT;
      }
    }

    // ---- online softmax over m (per query n = lane&15; pair via xor 16) ----
    float t = lg[0][0];
#pragma unroll
    for (int j = 1; j < 8; ++j) t = fmaxf(t, lg[0][j]);
#pragma unroll
    for (int j = 0; j < 8; ++j) t = fmaxf(t, lg[1][j]);
    t = fmaxf(t, __shfl_xor(t, 16));
    const float mnew = fmaxf(mrun, t);
    const float corr = __expf(mrun - mnew);
    mrun = mnew;

    float pr[2][8]; float ps = 0.f;
#pragma unroll
    for (int ms = 0; ms < 2; ++ms)
#pragma unroll
      for (int j = 0; j < 8; ++j) { pr[ms][j] = __expf(lg[ms][j] - mnew); ps += pr[ms][j]; }
    ps += __shfl_xor(ps, 16);
    lrun = lrun * corr + ps;

    // rescale accumulators: acc row (= query) 8*half+j needs corr of lane 8*half+j
    float c8[8];
#pragma unroll
    for (int j = 0; j < 8; ++j) c8[j] = __shfl(corr, 8 * half + j);
#pragma unroll
    for (int g = 0; g < 4; ++g)
#pragma unroll
      for (int j = 0; j < 8; ++j) acc[g][j] *= c8[j];

    // P already in A-frag layout: row n = ln; e<8 -> m=8*half+e (tile ms=0),
    // e>=8 -> m=16+8*half+(e-8) (tile ms=1)
    v16bf pa;
#pragma unroll
    for (int e = 0; e < 8; ++e) { pa[e] = (__bf16)pr[0][e]; pa[8 + e] = (__bf16)pr[1][e]; }

#pragma unroll
    for (int g = 0; g < 4; ++g) {
      const __bf16* vl = &vlds[p][(g * 16 + ln) * 36 + 8 * half];
      v16bf vf;
#pragma unroll
      for (int e = 0; e < 16; ++e) vf[e] = vl[e + (e >= 8 ? 8 : 0)];
      acc[g] = wmma_bf16(pa, vf, acc[g]);
    }

    if (nxt) vstore(p ^ 1, vreg);   // publish next V tile
    wait_async0();                  // drain async K DMA before barrier
    __syncthreads();
  }

  // ---- normalize and store O (bf16 workspace) ----
  const float inv = 1.0f / lrun;
  float i8[8];
#pragma unroll
  for (int j = 0; j < 8; ++j) i8[j] = __shfl(inv, 8 * half + j);
#pragma unroll
  for (int g = 0; g < 4; ++g)
#pragma unroll
    for (int j = 0; j < 8; ++j) {
      const int ng = q0 + wave * 16 + 8 * half + j;
      const int c  = k * HD + g * 16 + ln;
      O[(size_t)(b * SEQ + ng) * CDIM + c] = (__bf16)(acc[g][j] * i8[j]);
    }
}

// ---------------------------------------------------------------------------
extern "C" void kernel_launch(void* const* d_in, const int* in_sizes, int n_in,
                              void* d_out, int out_size, void* d_ws, size_t ws_size,
                              hipStream_t stream) {
  (void)in_sizes; (void)n_in; (void)out_size; (void)ws_size;
  const float* yp   = (const float*)d_in[0];  // [2,2048,1024]
  const float* mask = (const float*)d_in[1];  // [2048,2048]
  const float* Lx   = (const float*)d_in[2];  // [16,1024,64]
  const float* Ly   = (const float*)d_in[3];
  const float* Tx   = (const float*)d_in[4];
  const float* Ty   = (const float*)d_in[5];

  const size_t TSZ = (size_t)NROWS * CDIM;    // 4M bf16 = 8MB each
  __bf16* Rw = (__bf16*)d_ws;                 // keys    (x @ Lam_x)
  __bf16* Sw = Rw + TSZ;                      // queries (y' @ Lam_y)
  __bf16* Vw = Sw + TSZ;                      // values  (x @ Th_x)
  __bf16* Ow = Vw + TSZ;                      // attention output

  dim3 gg(NROWS / 128, CDIM / 64);
  gemm1024_wmma<0, float, __bf16><<<gg, 256, 0, stream>>>(yp, Lx, Rw);
  gemm1024_wmma<0, float, __bf16><<<gg, 256, 0, stream>>>(yp, Ly, Sw);
  gemm1024_wmma<0, float, __bf16><<<gg, 256, 0, stream>>>(yp, Tx, Vw);

  flash_attn_wmma<<<dim3(SEQ / 64, NHEAD, NB), 128, 0, stream>>>(Rw, Sw, Vw, mask, Ow);

  gemm1024_wmma<1, __bf16, float><<<gg, 256, 0, stream>>>(Ow, Ty, (float*)d_out);
}
